// API_RNNAgent_8186207666477
// MI455X (gfx1250) — matrix operand order repaired and checked by reading, not compiled, not run on hardware
//
#include <hip/hip_runtime.h>

typedef __attribute__((ext_vector_type(16))) __bf16 v16bf;
typedef __attribute__((ext_vector_type(8)))  __bf16 v8bf;
typedef __attribute__((ext_vector_type(8)))  float  v8f;

// ---- problem constants (fixed by the harness) ----
constexpr int BS   = 4096;
constexpr int NAG  = 8;
constexpr int NROW = BS * NAG;        // 32768 agent rows
constexpr int HD   = 64;              // H == HYP == 64
constexpr int NEN  = 8;
constexpr int NAL  = 7;
constexpr int EF   = 9;
constexpr int AF   = 10;
constexpr int OWNF = 16;
constexpr int R_EN = NROW * NEN;      // 262144 enemy rows
constexpr int R_AL = NROW * NAL;      // 229376 ally rows
constexpr int QCOLS = 14;
constexpr int QSIZE = NROW * QCOLS;   // q floats, then hh (NROW*64)

// ---- workspace layout (BYTE offsets; gi/gh alias dead emb regions) ----
constexpr size_t BO_RELU_E = 0;                                    // R_EN*64 f32
constexpr size_t BO_RELU_A = BO_RELU_E + (size_t)R_EN * HD * 4;    // R_AL*64 f32
constexpr size_t BO_EMB_EN = BO_RELU_A + (size_t)R_AL * HD * 4;    // R_EN*64 f32
constexpr size_t BO_EMB_AL = BO_EMB_EN + (size_t)R_EN * HD * 4;    // R_AL*64 f32
constexpr size_t BO_U      = BO_EMB_AL + (size_t)R_AL * HD * 4;    // NROW*64 f32
constexpr size_t BO_GI     = BO_EMB_EN;                            // NROW*192 f32 (alias)
constexpr size_t BO_GH     = BO_EMB_AL;                            // NROW*192 f32 (alias)
constexpr size_t BO_XBF    = BO_U + (size_t)NROW * HD * 4;         // NROW*64 bf16
constexpr size_t BO_HBF    = BO_XBF + (size_t)NROW * HD * 2;       // NROW*64 bf16
constexpr size_t BO_WTE    = BO_HBF + (size_t)NROW * HD * 2;       // 64*576 bf16
constexpr size_t BO_WTA    = BO_WTE + (size_t)HD * (EF * HD) * 2;  // 64*640 bf16
constexpr size_t BO_WIHB   = BO_WTA + (size_t)HD * (AF * HD) * 2;  // 192*64 bf16
constexpr size_t BO_WHHB   = BO_WIHB + (size_t)192 * HD * 2;       // 192*64 bf16

// ======================================================================
// Prep: f32 -> bf16 straight conversion
// ======================================================================
__global__ void cvt_bf16_kernel(const float* __restrict__ in,
                                __bf16* __restrict__ out, int n) {
  int idx = blockIdx.x * blockDim.x + threadIdx.x;
  if (idx < n) out[idx] = (__bf16)in[idx];
}

// ======================================================================
// Prep: reshape hypernet W2 [64, wstride] into Wt[n][kk] bf16,
//       kk = k*F + e, value = W2[k*wstride + e*64 + n].
// ======================================================================
template<int F>
__global__ void reshape_w2_kernel(const float* __restrict__ W2, int wstride,
                                  __bf16* __restrict__ Wt) {
  constexpr int KM = F * HD;
  int idx = blockIdx.x * blockDim.x + threadIdx.x;
  if (idx >= HD * KM) return;
  int n  = idx / KM;
  int kk = idx - n * KM;
  int k  = kk / F;
  int e  = kk - k * F;
  Wt[idx] = (__bf16)W2[(size_t)k * wstride + e * HD + n];
}

// ======================================================================
// Stage 1: relu(feats @ w1 + b1) for the hypernet first layers.
// ======================================================================
template<int F>
__global__ void relu1_kernel(const float* __restrict__ feats,
                             const float* __restrict__ w1,
                             const float* __restrict__ b1,
                             float* __restrict__ out, int R) {
  int idx = blockIdx.x * blockDim.x + threadIdx.x;
  if (idx >= R * HD) return;
  int row = idx >> 6;
  int j   = idx & 63;
  const float* f = feats + (size_t)row * F;
  float acc = b1[j];
#pragma unroll
  for (int e = 0; e < F; ++e) acc = fmaf(f[e], w1[e * HD + j], acc);
  out[idx] = fmaxf(acc, 0.0f);
}

// ======================================================================
// Stage 2: hypernet contraction GEMM over the rank-1 expansion.
// A row operands register-resident; B = one v16bf load per N-tile/step.
// Divergent A-build (literal `hi` per path) so r[k]/f[e] fold to static
// register indices after full unroll; wave reconverges before each WMMA.
// ======================================================================
template<int F>
__global__ void __launch_bounds__(256)
hyper_gemm_kernel(const float* __restrict__ relu_ws,
                  const float* __restrict__ feats,
                  const __bf16* __restrict__ Wt,   // [64, F*64]
                  float* __restrict__ out) {
  constexpr int KM = F * HD;
  constexpr int STEPS = KM / 32;

  const int lane = threadIdx.x & 31;
  const int wave = threadIdx.x >> 5;
  const bool hi  = lane >= 16;
  const int lnlo = lane & 15;
  const int m0   = (blockIdx.x * 8 + wave) * 16;
  const int mrow = m0 + lnlo;

  const float* rrow = relu_ws + (size_t)mrow * HD;
  const float* frow = feats   + (size_t)mrow * F;

  float r[HD];
#pragma unroll
  for (int k = 0; k < HD; ++k) r[k] = rrow[k];
  float f[F];
#pragma unroll
  for (int e = 0; e < F; ++e) f[e] = frow[e];

  v8f c[4];
#pragma unroll
  for (int j = 0; j < 4; ++j)
#pragma unroll
    for (int q = 0; q < 8; ++q) c[j][q] = 0.0f;

  const __bf16* wbase = Wt + (size_t)lnlo * KM + (hi ? 16 : 0);

#pragma unroll
  for (int ks = 0; ks < STEPS; ++ks) {
    const int kk0 = ks * 32;
    v16bf a;
    if (!hi) {
#pragma unroll
      for (int i = 0; i < 16; ++i) {
        int kk = kk0 + i;   // K = i
        int k  = kk / F;
        int e  = kk - k * F;
        a[i] = (__bf16)(r[k] * f[e]);
      }
    } else {
#pragma unroll
      for (int i = 0; i < 16; ++i) {
        int kk = kk0 + 16 + i;
        int k  = kk / F;
        int e  = kk - k * F;
        a[i] = (__bf16)(r[k] * f[e]);
      }
    }
    v16bf b0 = *(const v16bf*)(wbase + (size_t)0 * 16 * KM + kk0);
    v16bf b1 = *(const v16bf*)(wbase + (size_t)1 * 16 * KM + kk0);
    v16bf b2 = *(const v16bf*)(wbase + (size_t)2 * 16 * KM + kk0);
    v16bf b3 = *(const v16bf*)(wbase + (size_t)3 * 16 * KM + kk0);
    c[0] = __builtin_amdgcn_wmma_f32_16x16x32_bf16(false, a, false, b0, (short)0, c[0], false, false);
    c[1] = __builtin_amdgcn_wmma_f32_16x16x32_bf16(false, a, false, b1, (short)0, c[1], false, false);
    c[2] = __builtin_amdgcn_wmma_f32_16x16x32_bf16(false, a, false, b2, (short)0, c[2], false, false);
    c[3] = __builtin_amdgcn_wmma_f32_16x16x32_bf16(false, a, false, b3, (short)0, c[3], false, false);
  }

#pragma unroll
  for (int j = 0; j < 4; ++j) {
    int n = j * 16 + lnlo;
#pragma unroll
    for (int q = 0; q < 8; ++q) {
      int m = m0 + q + (hi ? 8 : 0);
      out[(size_t)m * HD + n] = c[j][q];
    }
  }
}

// ======================================================================
// Stage 3: x = relu(own@fc1 + embeddings + sum emb_en + sum emb_al
//                   + hypernet-b2 bias terms) -> bf16 for the GRU GEMM
// ======================================================================
__global__ void combine_x_kernel(const float* __restrict__ own,
                                 const float* __restrict__ enemy,
                                 const float* __restrict__ ally,
                                 const float* __restrict__ fc1w,
                                 const float* __restrict__ fc1b,
                                 const float* __restrict__ aemb,
                                 const float* __restrict__ actemb,
                                 const int*   __restrict__ aidx,
                                 const int*   __restrict__ lidx,
                                 const float* __restrict__ heb2,
                                 const float* __restrict__ hab2,
                                 const float* __restrict__ embEn,
                                 const float* __restrict__ embAl,
                                 __bf16* __restrict__ x) {
  int idx = blockIdx.x * blockDim.x + threadIdx.x;
  if (idx >= NROW * HD) return;
  int row = idx >> 6;
  int h   = idx & 63;

  float acc = fc1b[h];
  const float* orow = own + (size_t)row * OWNF;
#pragma unroll
  for (int ff = 0; ff < OWNF; ++ff) acc = fmaf(orow[ff], fc1w[ff * HD + h], acc);
  acc += aemb[aidx[row] * HD + h] + actemb[lidx[row] * HD + h];

  for (int e = 0; e < NEN; ++e) {
    size_t er = (size_t)row * NEN + e;
    acc += embEn[er * HD + h];
    const float* ef = enemy + er * EF;
#pragma unroll
    for (int ff = 0; ff < EF; ++ff) acc = fmaf(ef[ff], heb2[ff * HD + h], acc);
  }
  for (int al = 0; al < NAL; ++al) {
    size_t ar = (size_t)row * NAL + al;
    acc += embAl[ar * HD + h];
    const float* af = ally + ar * AF;
#pragma unroll
    for (int ff = 0; ff < AF; ++ff) acc = fmaf(af[ff], hab2[ff * HD + h], acc);
  }
  x[idx] = (__bf16)fmaxf(acc, 0.0f);
}

// ======================================================================
// Stage 4: GRU gate GEMM: out[M,192] = A[M,64] @ W[192,64]^T + bias
// A, W pre-converted bf16: vector fragment loads only.
// ======================================================================
__global__ void __launch_bounds__(256)
gru_gemm_kernel(const __bf16* __restrict__ Abf,
                const __bf16* __restrict__ Wb,
                const float* __restrict__ bias,
                float* __restrict__ out) {
  const int lane = threadIdx.x & 31;
  const int wave = threadIdx.x >> 5;
  const bool hi  = lane >= 16;
  const int lnlo = lane & 15;
  const int m0   = (blockIdx.x * 8 + wave) * 16;
  const __bf16* ap = Abf + (size_t)(m0 + lnlo) * HD + (hi ? 16 : 0);

  v8bf l0 = *(const v8bf*)(ap);        // K in [hi*16, hi*16+8)
  v8bf l1 = *(const v8bf*)(ap + 8);    // K in [hi*16+8, hi*16+16)
  v8bf l2 = *(const v8bf*)(ap + 32);
  v8bf l3 = *(const v8bf*)(ap + 40);
  v16bf a0, a1;
#pragma unroll
  for (int i = 0; i < 8; ++i) {
    a0[i] = l0[i]; a0[8 + i] = l1[i];
    a1[i] = l2[i]; a1[8 + i] = l3[i];
  }

#pragma unroll
  for (int j = 0; j < 12; ++j) {        // N = 192
    int n = j * 16 + lnlo;
    const __bf16* wrow = Wb + (size_t)n * HD + (hi ? 16 : 0);
    v16bf b0 = *(const v16bf*)(wrow);
    v16bf b1 = *(const v16bf*)(wrow + 32);
    v8f c;
#pragma unroll
    for (int q = 0; q < 8; ++q) c[q] = 0.0f;
    c = __builtin_amdgcn_wmma_f32_16x16x32_bf16(false, a0, false, b0, (short)0, c, false, false);
    c = __builtin_amdgcn_wmma_f32_16x16x32_bf16(false, a1, false, b1, (short)0, c, false, false);

    float bb = bias[n];
#pragma unroll
    for (int q = 0; q < 8; ++q) {
      int m = m0 + q + (hi ? 8 : 0);
      out[(size_t)m * 192 + n] = c[q] + bb;
    }
  }
}

// ======================================================================
// Stage 5: elementwise GRU cell -> hh
// ======================================================================
__global__ void gru_elem_kernel(const float* __restrict__ gi,
                                const float* __restrict__ gh,
                                const float* __restrict__ h_in,
                                float* __restrict__ hh) {
  int idx = blockIdx.x * blockDim.x + threadIdx.x;
  if (idx >= NROW * HD) return;
  int row = idx >> 6;
  int h   = idx & 63;
  size_t base = (size_t)row * 192;
  float ir = gi[base + h],       hr = gh[base + h];
  float iz = gi[base + 64 + h],  hz = gh[base + 64 + h];
  float in = gi[base + 128 + h], hn = gh[base + 128 + h];
  float r = 1.0f / (1.0f + __expf(-(ir + hr)));
  float z = 1.0f / (1.0f + __expf(-(iz + hz)));
  float n = tanhf(in + r * hn);
  hh[idx] = (1.0f - z) * n + z * h_in[idx];
}

// ======================================================================
// Stage 6: u[row,k] = sum_h he_w2[k,576+h]*hh[row,h] + he_w2[k,640]
// ======================================================================
__global__ void u_kernel(const float* __restrict__ hew2,
                         const float* __restrict__ hh,
                         float* __restrict__ u) {
  int idx = blockIdx.x * blockDim.x + threadIdx.x;
  if (idx >= NROW * HD) return;
  int row = idx >> 6;
  int k   = idx & 63;
  const float* wrow = hew2 + (size_t)k * 641 + 576;
  const float* hrow = hh + (size_t)row * HD;
  float acc = wrow[64];                 // he_w2[k, 640]
#pragma unroll
  for (int h = 0; h < HD; ++h) acc = fmaf(wrow[h], hrow[h], acc);
  u[idx] = acc;
}

// ======================================================================
// Stage 7: q_normal (fc2) + q_attack = relu_e . u + bias terms
// ======================================================================
__global__ void q_kernel(const float* __restrict__ hh,
                         const float* __restrict__ fc2w,
                         const float* __restrict__ fc2b,
                         const float* __restrict__ heb2,
                         const float* __restrict__ relu_e,
                         const float* __restrict__ u,
                         float* __restrict__ q) {
  int idx = blockIdx.x * blockDim.x + threadIdx.x;
  if (idx >= NROW * QCOLS) return;
  int row = idx / QCOLS;
  int o   = idx - row * QCOLS;
  const float* hrow = hh + (size_t)row * HD;
  float acc;
  if (o < 6) {
    acc = fc2b[o];
#pragma unroll
    for (int h = 0; h < HD; ++h) acc = fmaf(hrow[h], fc2w[h * 6 + o], acc);
  } else {
    int e = o - 6;
    float s = heb2[640];
#pragma unroll
    for (int h = 0; h < HD; ++h) s = fmaf(hrow[h], heb2[576 + h], s);
    acc = s;
    const float* rrow = relu_e + ((size_t)row * NEN + e) * HD;
    const float* urow = u + (size_t)row * HD;
#pragma unroll
    for (int k = 0; k < HD; ++k) acc = fmaf(rrow[k], urow[k], acc);
  }
  q[idx] = acc;
}

// ======================================================================
extern "C" void kernel_launch(void* const* d_in, const int* in_sizes, int n_in,
                              void* d_out, int out_size, void* d_ws, size_t ws_size,
                              hipStream_t stream) {
  (void)in_sizes; (void)n_in; (void)out_size; (void)ws_size;
  const float* own    = (const float*)d_in[0];
  const float* enemy  = (const float*)d_in[1];
  const float* ally   = (const float*)d_in[2];
  const float* hidden = (const float*)d_in[3];
  const int*   aidx   = (const int*)d_in[4];
  const int*   lidx   = (const int*)d_in[5];
  const float* fc1w   = (const float*)d_in[7];
  const float* fc1b   = (const float*)d_in[8];
  const float* aemb   = (const float*)d_in[9];
  const float* actemb = (const float*)d_in[10];
  const float* hew1   = (const float*)d_in[11];
  const float* heb1   = (const float*)d_in[12];
  const float* hew2   = (const float*)d_in[13];
  const float* heb2   = (const float*)d_in[14];
  const float* haw1   = (const float*)d_in[15];
  const float* hab1   = (const float*)d_in[16];
  const float* haw2   = (const float*)d_in[17];
  const float* hab2   = (const float*)d_in[18];
  const float* wih    = (const float*)d_in[19];
  const float* whh    = (const float*)d_in[20];
  const float* bih    = (const float*)d_in[21];
  const float* bhh    = (const float*)d_in[22];
  const float* fc2w   = (const float*)d_in[23];
  const float* fc2b   = (const float*)d_in[24];

  char*  wsb = (char*)d_ws;
  float* out = (float*)d_out;
  float* hhp = out + QSIZE;

  float*  reluE = (float*)(wsb + BO_RELU_E);
  float*  reluA = (float*)(wsb + BO_RELU_A);
  float*  embEn = (float*)(wsb + BO_EMB_EN);
  float*  embAl = (float*)(wsb + BO_EMB_AL);
  float*  uW    = (float*)(wsb + BO_U);
  float*  gi    = (float*)(wsb + BO_GI);
  float*  gh    = (float*)(wsb + BO_GH);
  __bf16* xbf   = (__bf16*)(wsb + BO_XBF);
  __bf16* hbf   = (__bf16*)(wsb + BO_HBF);
  __bf16* wtE   = (__bf16*)(wsb + BO_WTE);
  __bf16* wtA   = (__bf16*)(wsb + BO_WTA);
  __bf16* wihB  = (__bf16*)(wsb + BO_WIHB);
  __bf16* whhB  = (__bf16*)(wsb + BO_WHHB);

  // 0: weight reshape / bf16 conversions (tiny, L2-resident afterwards)
  reshape_w2_kernel<EF><<<(HD * EF * HD + 255) / 256, 256, 0, stream>>>(
      hew2, (EF + 1) * HD + 1, wtE);
  reshape_w2_kernel<AF><<<(HD * AF * HD + 255) / 256, 256, 0, stream>>>(
      haw2, AF * HD, wtA);
  cvt_bf16_kernel<<<(192 * HD + 255) / 256, 256, 0, stream>>>(wih, wihB, 192 * HD);
  cvt_bf16_kernel<<<(192 * HD + 255) / 256, 256, 0, stream>>>(whh, whhB, 192 * HD);
  cvt_bf16_kernel<<<(NROW * HD + 255) / 256, 256, 0, stream>>>(hidden, hbf, NROW * HD);

  // 1: hypernet layer-1 relu
  relu1_kernel<EF><<<(R_EN * HD + 255) / 256, 256, 0, stream>>>(
      enemy, hew1, heb1, reluE, R_EN);
  relu1_kernel<AF><<<(R_AL * HD + 255) / 256, 256, 0, stream>>>(
      ally, haw1, hab1, reluA, R_AL);

  // 2: WMMA hypernet contraction GEMMs (wave = 16 rows x 64 cols)
  hyper_gemm_kernel<EF><<<R_EN / 128, 256, 0, stream>>>(reluE, enemy, wtE, embEn);
  hyper_gemm_kernel<AF><<<R_AL / 128, 256, 0, stream>>>(reluA, ally, wtA, embAl);

  // 3: combine + relu -> x (bf16)
  combine_x_kernel<<<(NROW * HD + 255) / 256, 256, 0, stream>>>(
      own, enemy, ally, fc1w, fc1b, aemb, actemb, aidx, lidx,
      heb2, hab2, embEn, embAl, xbf);

  // 4: WMMA GRU gate GEMMs
  gru_gemm_kernel<<<NROW / 128, 256, 0, stream>>>(xbf, wihB, bih, gi);
  gru_gemm_kernel<<<NROW / 128, 256, 0, stream>>>(hbf, whhB, bhh, gh);

  // 5: GRU cell -> hh (second output)
  gru_elem_kernel<<<(NROW * HD + 255) / 256, 256, 0, stream>>>(gi, gh, hidden, hhp);

  // 6: attack-head weight contraction u = W2_tail @ hh
  u_kernel<<<(NROW * HD + 255) / 256, 256, 0, stream>>>(hew2, hhp, uW);

  // 7: q heads -> first output
  q_kernel<<<(NROW * QCOLS + 255) / 256, 256, 0, stream>>>(
      hhp, fc2w, fc2b, heb2, reluE, uW, out);
}
